// DAWNLayer_46016279610309
// MI455X (gfx1250) — matrix operand order repaired and checked by reading, not compiled, not run on hardware
//
#include <hip/hip_runtime.h>
#include <hip/hip_bf16.h>

// ---------------------------------------------------------------------------
// DAWN layer for MI455X (gfx1250): bf16 WMMA GEMMs (LDS double-buffered via
// async global->LDS copies), flash attention, router.
// ---------------------------------------------------------------------------

typedef __bf16 bf16;
typedef __bf16 bf16x8  __attribute__((ext_vector_type(8)));
typedef __bf16 bf16x16 __attribute__((ext_vector_type(16)));
typedef float  f32x8   __attribute__((ext_vector_type(8)));
typedef int    int4v   __attribute__((ext_vector_type(4)));
typedef int    v4i_gcc __attribute__((__vector_size__(4 * sizeof(int))));

#define WMMA_BF16(a, b, c) \
  __builtin_amdgcn_wmma_f32_16x16x32_bf16(false, (a), false, (b), (short)0, (c), false, false)

// ---- async global -> LDS (CDNA5), with safe synchronous fallback ----------
#if defined(__has_builtin)
#if __has_builtin(__builtin_amdgcn_global_load_async_to_lds_b128)
#define HAVE_ASYNC_LDS 1
#endif
#endif
#ifndef HAVE_ASYNC_LDS
#define HAVE_ASYNC_LDS 0
#endif

__device__ __forceinline__ void copy16_g2l(bf16* dst_lds, const bf16* src) {
#if HAVE_ASYNC_LDS
  typedef __attribute__((address_space(1))) v4i_gcc* gptr_t;
  typedef __attribute__((address_space(3))) v4i_gcc* lptr_t;
  __builtin_amdgcn_global_load_async_to_lds_b128(
      (gptr_t)(void*)const_cast<bf16*>(src),
      (lptr_t)(void*)dst_lds, 0, 0);
#else
  *(int4v*)dst_lds = *(const int4v*)src;
#endif
}

#if HAVE_ASYNC_LDS
#define ASYNC_WAIT(n) asm volatile("s_wait_asynccnt " #n ::: "memory")
#else
#define ASYNC_WAIT(n) do {} while (0)
#endif

namespace cfg {
constexpr int B   = 2;
constexpr int S   = 2048;
constexpr int D   = 1024;
constexpr int DFF = 4096;
constexpr int H   = 16;
constexpr int DH  = 64;
constexpr int NN  = 256;
constexpr int TOPK = 8;
constexpr int BS  = B * S;
constexpr float EPS = 1e-5f;
}

enum Epi { EPI_NONE = 0, EPI_RESADD = 1, EPI_GELU = 2 };

union BFrag {
  bf16x16 v;
  bf16x8  h[2];
};

// ---------------------------------------------------------------------------
// fp32 -> bf16 conversion (weights)
// ---------------------------------------------------------------------------
__global__ void __launch_bounds__(256) cvt_f32_bf16(const float* __restrict__ src,
                                                    bf16* __restrict__ dst, int n) {
  int i = blockIdx.x * 256 + threadIdx.x;
  if (i < n) dst[i] = (bf16)src[i];
}

// ---------------------------------------------------------------------------
// Fused dual LayerNorm: n1 (bf16, into comb[:, 0:D]) and n2 (f32)
// ---------------------------------------------------------------------------
__global__ void __launch_bounds__(256) ln_dual_kernel(
    const float* __restrict__ x,
    const float* __restrict__ g1, const float* __restrict__ b1,
    const float* __restrict__ g2, const float* __restrict__ b2,
    bf16* __restrict__ comb, float* __restrict__ n2) {
  using namespace cfg;
  const int row = blockIdx.x;
  const int tid = threadIdx.x;
  const float* xr = x + (size_t)row * D;

  float xv[4];
  float s = 0.f, ss = 0.f;
#pragma unroll
  for (int i = 0; i < 4; ++i) {
    xv[i] = xr[tid + i * 256];
    s += xv[i];
    ss += xv[i] * xv[i];
  }
#pragma unroll
  for (int off = 16; off >= 1; off >>= 1) {
    s  += __shfl_xor(s, off, 32);
    ss += __shfl_xor(ss, off, 32);
  }
  __shared__ float sh_s[8], sh_ss[8];
  if ((tid & 31) == 0) { sh_s[tid >> 5] = s; sh_ss[tid >> 5] = ss; }
  __syncthreads();
  float tot = 0.f, tot2 = 0.f;
#pragma unroll
  for (int i = 0; i < 8; ++i) { tot += sh_s[i]; tot2 += sh_ss[i]; }
  const float mean = tot / (float)D;
  const float var  = tot2 / (float)D - mean * mean;
  const float inv  = rsqrtf(var + EPS);

#pragma unroll
  for (int i = 0; i < 4; ++i) {
    const int c = tid + i * 256;
    const float xn = (xv[i] - mean) * inv;
    comb[(size_t)row * (2 * D) + c] = (bf16)(xn * g1[c] + b1[c]);
    n2[(size_t)row * D + c]         = xn * g2[c] + b2[c];
  }
}

// ---------------------------------------------------------------------------
// GEMM: Y[M,N] = A[M,K](bf16, ld=lda) @ W[N,K]^T (+bias)(+epi)
// 256 threads, block tile 128x128, BK=32, double-buffered LDS staging via
// async global->LDS. Wave tile 32(M) x 64(N). All dims multiples of tiles.
// LDS rows padded (+8 elems) so 16-lane ds_load_b128 are bank-conflict-free.
// ---------------------------------------------------------------------------
template <bool OUT_BF16, int EPI>
__global__ void __launch_bounds__(256) gemm_bf16_kernel(
    const bf16* __restrict__ A, int lda,
    const bf16* __restrict__ W, int ldw,
    const float* __restrict__ bias,       // [N] or nullptr
    const float* __restrict__ res,        // [M,N] row-major (ld=N) or nullptr
    void* __restrict__ out, int ldo,
    int M, int N, int K) {
  constexpr int LD = 40;  // 32 + 8 pad
  __shared__ __align__(16) bf16 Ash[2][128][LD];
  __shared__ __align__(16) bf16 Wsh[2][128][LD];

  const int tid   = threadIdx.x;
  const int lane  = tid & 31;
  const int w     = tid >> 5;
  const int wm    = w & 3;    // 4 waves along M
  const int wn    = w >> 2;   // 2 waves along N
  const int laneM = lane & 15;
  const int laneH = lane >> 4;

  const int blockM = blockIdx.y * 128;
  const int blockN = blockIdx.x * 128;

  const bf16* gA = A + (size_t)blockM * lda;
  const bf16* gW = W + (size_t)blockN * ldw;

  // 128 rows x 32 cols x 2B = 8KB per tensor = 512 x 16B chunks; 2 per thread
  auto stage = [&](int k0, int buf) {
#pragma unroll
    for (int i = 0; i < 2; ++i) {
      const int chunk = tid + 256 * i;       // 0..511
      const int row   = chunk >> 2;          // 4 chunks per row
      const int col   = (chunk & 3) * 8;     // elements
      copy16_g2l(&Ash[buf][row][col], gA + (size_t)row * lda + k0 + col);
      copy16_g2l(&Wsh[buf][row][col], gW + (size_t)row * ldw + k0 + col);
    }
  };

  f32x8 acc[2][4] = {};
  const int nk = K >> 5;
  stage(0, 0);
  int buf = 0;
  for (int t = 0; t < nk; ++t) {
    if (t + 1 < nk) {
      stage((t + 1) << 5, buf ^ 1);  // prefetch next stage (4 async/thread)
      ASYNC_WAIT(4);                 // current stage retired, next in flight
    } else {
      ASYNC_WAIT(0);
    }
    __syncthreads();

    BFrag a[2], b[4];
#pragma unroll
    for (int mt = 0; mt < 2; ++mt) {
      const bf16* r = &Ash[buf][wm * 32 + mt * 16 + laneM][0];
      a[mt].h[0] = *(const bf16x8*)(r + 8 * laneH);
      a[mt].h[1] = *(const bf16x8*)(r + 16 + 8 * laneH);
    }
#pragma unroll
    for (int nt = 0; nt < 4; ++nt) {
      const bf16* r = &Wsh[buf][wn * 64 + nt * 16 + laneM][0];
      b[nt].h[0] = *(const bf16x8*)(r + 8 * laneH);
      b[nt].h[1] = *(const bf16x8*)(r + 16 + 8 * laneH);
    }
#pragma unroll
    for (int mt = 0; mt < 2; ++mt)
#pragma unroll
      for (int nt = 0; nt < 4; ++nt)
        acc[mt][nt] = WMMA_BF16(a[mt].v, b[nt].v, acc[mt][nt]);

    __syncthreads();
    buf ^= 1;
  }

  // Epilogue. C layout: row = laneH*8 + v, col = laneM (within each 16x16).
#pragma unroll
  for (int mt = 0; mt < 2; ++mt) {
#pragma unroll
    for (int nt = 0; nt < 4; ++nt) {
      const int n = blockN + wn * 64 + nt * 16 + laneM;
      const float bn = bias ? bias[n] : 0.f;
#pragma unroll
      for (int v = 0; v < 8; ++v) {
        const int m = blockM + wm * 32 + mt * 16 + laneH * 8 + v;
        float val = acc[mt][nt][v] + bn;
        if constexpr (EPI == EPI_GELU)
          val = 0.5f * val * (1.0f + erff(val * 0.70710678118654752f));
        if constexpr (EPI == EPI_RESADD)
          val += res[(size_t)m * N + n];
        if constexpr (OUT_BF16)
          ((bf16*)out)[(size_t)m * ldo + n] = (bf16)val;
        else
          ((float*)out)[(size_t)m * ldo + n] = val;
      }
    }
  }
}

// ---------------------------------------------------------------------------
// V transpose: v [B,S,H,DH] bf16 -> vt [B,H,DH,S] bf16
// ---------------------------------------------------------------------------
__global__ void __launch_bounds__(256) transpose_v_kernel(
    const bf16* __restrict__ v, bf16* __restrict__ vt) {
  using namespace cfg;
  const int i = blockIdx.x * 256 + threadIdx.x;  // over BS*D
  const int d    = i & (DH - 1);
  const int h    = (i >> 6) & (H - 1);
  const int srow = i >> 10;            // b*S + s
  const int b    = srow >> 11;         // / S
  const int sIdx = srow & (S - 1);
  vt[((size_t)(b * H + h) * DH + d) * S + sIdx] = v[i];
}

// ---------------------------------------------------------------------------
// Flash attention (bf16 WMMA, online softmax).
// grid = (B*H, S/128). 256 threads; each wave owns 16 query rows.
// K / V^T tiles staged in LDS once per block (all 8 waves share), double
// buffered via async global->LDS. Writes ctx (bf16) into comb[:, D:2D].
// ---------------------------------------------------------------------------
__global__ void __launch_bounds__(256) attn_kernel(
    const bf16* __restrict__ q, const bf16* __restrict__ k,
    const bf16* __restrict__ vt, bf16* __restrict__ comb) {
  using namespace cfg;
  constexpr int LD = 72;  // 64 + 8 pad
  __shared__ __align__(16) bf16 Ksh[2][64][LD];
  __shared__ __align__(16) bf16 Vsh[2][64][LD];
  __shared__ __align__(16) bf16 Plds[8][16][64];

  const int bh   = blockIdx.x;
  const int b    = bh / H;
  const int h    = bh % H;
  const int tid  = threadIdx.x;
  const int lane = tid & 31;
  const int w    = tid >> 5;
  const int laneM = lane & 15;
  const int laneH = lane >> 4;
  const int q0 = blockIdx.y * 128 + w * 16;

  const bf16* kbase  = k + (size_t)(b * S) * D + h * DH;
  const bf16* vtbase = vt + (size_t)bh * DH * S;

  // 64 rows x 64 cols x 2B = 8KB per tensor = 512 x 16B chunks; 2 per thread
  auto stage = [&](int j0, int buf) {
#pragma unroll
    for (int i = 0; i < 2; ++i) {
      const int chunk = tid + 256 * i;       // 0..511
      const int row   = chunk >> 3;          // 8 chunks per row
      const int col   = (chunk & 7) * 8;     // elements
      copy16_g2l(&Ksh[buf][row][col], kbase + (size_t)(j0 + row) * D + col);
      copy16_g2l(&Vsh[buf][row][col], vtbase + (size_t)row * S + j0 + col);
    }
  };

  // Q fragments (A-layout), rows q0+laneM, head dim 0..63 (two K=32 frags)
  BFrag qa[2];
  {
    const bf16* qrow = q + (size_t)(b * S + q0 + laneM) * D + h * DH;
#pragma unroll
    for (int kt = 0; kt < 2; ++kt) {
      qa[kt].h[0] = *(const bf16x8*)(qrow + kt * 32 + 8 * laneH);
      qa[kt].h[1] = *(const bf16x8*)(qrow + kt * 32 + 16 + 8 * laneH);
    }
  }

  f32x8 o[4] = {};
  float rm[8], rl[8];
#pragma unroll
  for (int v = 0; v < 8; ++v) { rm[v] = -1e30f; rl[v] = 0.f; }

  const float scale = 0.125f;  // 1/sqrt(DH)

  stage(0, 0);
  int buf = 0;
  for (int j0 = 0; j0 < S; j0 += 64) {
    if (j0 + 64 < S) {
      stage(j0 + 64, buf ^ 1);
      ASYNC_WAIT(4);
    } else {
      ASYNC_WAIT(0);
    }
    __syncthreads();

    // ---- S = Q K^T  (16 x 64 tile per wave), K from LDS ----
    f32x8 sacc[4] = {};
#pragma unroll
    for (int nt = 0; nt < 4; ++nt) {
      const bf16* krow = &Ksh[buf][nt * 16 + laneM][0];
#pragma unroll
      for (int kt = 0; kt < 2; ++kt) {
        BFrag kb;
        kb.h[0] = *(const bf16x8*)(krow + kt * 32 + 8 * laneH);
        kb.h[1] = *(const bf16x8*)(krow + kt * 32 + 16 + 8 * laneH);
        sacc[nt] = WMMA_BF16(qa[kt].v, kb.v, sacc[nt]);
      }
    }

    // ---- online softmax; butterfly steps batched across all 8 rows ----
    float mx[8], rs[8], alpha[8];
#pragma unroll
    for (int v = 0; v < 8; ++v)
      mx[v] = scale * fmaxf(fmaxf(sacc[0][v], sacc[1][v]),
                            fmaxf(sacc[2][v], sacc[3][v]));
#pragma unroll
    for (int off = 8; off >= 1; off >>= 1)
#pragma unroll
      for (int v = 0; v < 8; ++v)
        mx[v] = fmaxf(mx[v], __shfl_xor(mx[v], off, 32));
#pragma unroll
    for (int v = 0; v < 8; ++v) {
      const float mnew = fmaxf(rm[v], mx[v]);
      alpha[v] = __expf(rm[v] - mnew);
      rm[v] = mnew;
      float r = 0.f;
#pragma unroll
      for (int nt = 0; nt < 4; ++nt) {
        const float p = __expf(sacc[nt][v] * scale - mnew);
        sacc[nt][v] = p;
        r += p;
      }
      rs[v] = r;
    }
#pragma unroll
    for (int off = 8; off >= 1; off >>= 1)
#pragma unroll
      for (int v = 0; v < 8; ++v)
        rs[v] += __shfl_xor(rs[v], off, 32);
#pragma unroll
    for (int v = 0; v < 8; ++v) {
      rl[v] = rl[v] * alpha[v] + rs[v];
#pragma unroll
      for (int dt = 0; dt < 4; ++dt) o[dt][v] *= alpha[v];
    }

    // ---- P: C-layout regs -> LDS -> A-layout frags (per-wave region) ----
#pragma unroll
    for (int nt = 0; nt < 4; ++nt)
#pragma unroll
      for (int v = 0; v < 8; ++v)
        Plds[w][laneH * 8 + v][nt * 16 + laneM] = (bf16)sacc[nt][v];
    asm volatile("s_wait_dscnt 0x0" ::: "memory");
    BFrag pa[2];
#pragma unroll
    for (int kt = 0; kt < 2; ++kt) {
      pa[kt].h[0] = *(const bf16x8*)(&Plds[w][laneM][kt * 32 + 8 * laneH]);
      pa[kt].h[1] = *(const bf16x8*)(&Plds[w][laneM][kt * 32 + 16 + 8 * laneH]);
    }

    // ---- O += P @ V, V^T from LDS ----
#pragma unroll
    for (int dt = 0; dt < 4; ++dt) {
      const bf16* vrow = &Vsh[buf][dt * 16 + laneM][0];
#pragma unroll
      for (int kt = 0; kt < 2; ++kt) {
        BFrag vb;
        vb.h[0] = *(const bf16x8*)(vrow + kt * 32 + 8 * laneH);
        vb.h[1] = *(const bf16x8*)(vrow + kt * 32 + 16 + 8 * laneH);
        o[dt] = WMMA_BF16(pa[kt].v, vb.v, o[dt]);
      }
    }

    __syncthreads();
    buf ^= 1;
  }

  // ---- write ctx into comb[:, D:2D] ----
#pragma unroll
  for (int dt = 0; dt < 4; ++dt)
#pragma unroll
    for (int v = 0; v < 8; ++v) {
      const int m = q0 + laneH * 8 + v;
      const float val = o[dt][v] / rl[v];
      comb[(size_t)(b * S + m) * (2 * D) + D + h * DH + dt * 16 + laneM] = (bf16)val;
    }
}

// ---------------------------------------------------------------------------
// Router: per row, top-8 of 256 scores, softmax, weighted gather of emb (f32)
// ---------------------------------------------------------------------------
__global__ void __launch_bounds__(256) router_kernel(
    const float* __restrict__ scores,  // [BS, NN]
    const float* __restrict__ emb,     // [NN, D] f32
    bf16* __restrict__ info) {         // [BS, D]
  using namespace cfg;
  const int row = blockIdx.x;
  const int tid = threadIdx.x;
  __shared__ float shw[TOPK];
  __shared__ int   shi[TOPK];

  if (tid < 32) {
    const float* sr = scores + (size_t)row * NN;
    float vals[8];
#pragma unroll
    for (int i = 0; i < 8; ++i) vals[i] = sr[i * 32 + tid];
    float topv[TOPK];
    int   topi[TOPK];
#pragma unroll
    for (int t = 0; t < TOPK; ++t) {
      float lv = -3.0e38f;
      int   li = 0x7fffffff;
#pragma unroll
      for (int i = 0; i < 8; ++i)
        if (vals[i] > lv) { lv = vals[i]; li = i * 32 + tid; }
#pragma unroll
      for (int off = 16; off >= 1; off >>= 1) {
        const float ov = __shfl_xor(lv, off, 32);
        const int   oi = __shfl_xor(li, off, 32);
        if (ov > lv || (ov == lv && oi < li)) { lv = ov; li = oi; }
      }
      topv[t] = lv;
      topi[t] = li;
      if ((li & 31) == tid) vals[li >> 5] = -3.1e38f;  // knock out winner
    }
    if (tid < TOPK) {
      float sum = 0.f;
#pragma unroll
      for (int t = 0; t < TOPK; ++t) sum += __expf(topv[t] - topv[0]);
      shw[tid] = __expf(topv[tid] - topv[0]) / sum;
      shi[tid] = topi[tid];
    }
  }
  __syncthreads();

  float w8[TOPK];
  int   i8[TOPK];
#pragma unroll
  for (int t = 0; t < TOPK; ++t) { w8[t] = shw[t]; i8[t] = shi[t]; }
#pragma unroll
  for (int i = 0; i < 4; ++i) {
    const int d = tid + i * 256;
    float acc = 0.f;
#pragma unroll
    for (int t = 0; t < TOPK; ++t) acc += w8[t] * emb[(size_t)i8[t] * D + d];
    info[(size_t)row * D + d] = (bf16)acc;
  }
}

// ---------------------------------------------------------------------------
// Host orchestration
// ---------------------------------------------------------------------------
extern "C" void kernel_launch(void* const* d_in, const int* in_sizes, int n_in,
                              void* d_out, int out_size, void* d_ws, size_t ws_size,
                              hipStream_t stream) {
  using namespace cfg;
  const float* x    = (const float*)d_in[0];
  const float* Wq   = (const float*)d_in[1];
  const float* bq   = (const float*)d_in[2];
  const float* Wk   = (const float*)d_in[3];
  const float* bk   = (const float*)d_in[4];
  const float* Wv   = (const float*)d_in[5];
  const float* bv   = (const float*)d_in[6];
  const float* Wsp  = (const float*)d_in[7];
  const float* bsb  = (const float*)d_in[8];
  const float* emb  = (const float*)d_in[9];
  const float* Wnp  = (const float*)d_in[10];
  const float* bnp  = (const float*)d_in[11];
  const float* Wup  = (const float*)d_in[12];
  const float* bup  = (const float*)d_in[13];
  const float* Wdn  = (const float*)d_in[14];
  const float* bdn  = (const float*)d_in[15];
  const float* g1   = (const float*)d_in[16];
  const float* b1   = (const float*)d_in[17];
  const float* g2   = (const float*)d_in[18];
  const float* b2   = (const float*)d_in[19];
  float* out = (float*)d_out;

  char* cur = (char*)d_ws;
  auto alloc = [&](size_t bytes) -> char* {
    char* p = cur;
    cur += (bytes + 255) & ~(size_t)255;
    return p;
  };

  bf16* wq_b  = (bf16*)alloc((size_t)D * D * 2);
  bf16* wk_b  = (bf16*)alloc((size_t)D * D * 2);
  bf16* wv_b  = (bf16*)alloc((size_t)D * D * 2);
  bf16* ws_b  = (bf16*)alloc((size_t)D * 2 * D * 2);
  bf16* emb_b = (bf16*)alloc((size_t)NN * D * 2);
  bf16* wnp_b = (bf16*)alloc((size_t)D * D * 2);
  bf16* wup_b = (bf16*)alloc((size_t)DFF * D * 2);
  bf16* wdn_b = (bf16*)alloc((size_t)D * DFF * 2);
  bf16* comb  = (bf16*)alloc((size_t)BS * 2 * D * 2);  // [n1 | ctx] bf16
  float* n2f  = (float*)alloc((size_t)BS * D * 4);
  bf16* qb    = (bf16*)alloc((size_t)BS * D * 2);      // q,k,v,vt contiguous:
  bf16* kb    = (bf16*)alloc((size_t)BS * D * 2);      //   region reused for h
  bf16* vb    = (bf16*)alloc((size_t)BS * D * 2);
  bf16* vtb   = (bf16*)alloc((size_t)BS * D * 2);
  bf16* qryb  = (bf16*)alloc((size_t)BS * D * 2);
  float* scf  = (float*)alloc((size_t)BS * NN * 4);
  bf16* infob = (bf16*)alloc((size_t)BS * D * 2);
  bf16* xaugb = (bf16*)alloc((size_t)BS * D * 2);
  bf16* hb    = qb;  // 32 MB FFN hidden reuses q..vt (dead after attention)

  auto cvt = [&](const float* s, bf16* dp, int n) {
    cvt_f32_bf16<<<(n + 255) / 256, 256, 0, stream>>>(s, dp, n);
  };
  cvt(Wq,  wq_b,  D * D);
  cvt(Wk,  wk_b,  D * D);
  cvt(Wv,  wv_b,  D * D);
  cvt(Wsp, ws_b,  D * 2 * D);
  cvt(emb, emb_b, NN * D);
  cvt(Wnp, wnp_b, D * D);
  cvt(Wup, wup_b, DFF * D);
  cvt(Wdn, wdn_b, D * DFF);

  ln_dual_kernel<<<BS, 256, 0, stream>>>(x, g1, b1, g2, b2, comb, n2f);

  auto gemm = [&](const bf16* A, int lda, const bf16* W, int ldw,
                  const float* bias, const float* res, void* op, int ldo,
                  int M, int N, int K, int epi, bool outbf) {
    dim3 g(N / 128, M / 128), blk(256);
    if (outbf) {
      if (epi == EPI_GELU)
        gemm_bf16_kernel<true, EPI_GELU><<<g, blk, 0, stream>>>(A, lda, W, ldw, bias, res, op, ldo, M, N, K);
      else if (epi == EPI_RESADD)
        gemm_bf16_kernel<true, EPI_RESADD><<<g, blk, 0, stream>>>(A, lda, W, ldw, bias, res, op, ldo, M, N, K);
      else
        gemm_bf16_kernel<true, EPI_NONE><<<g, blk, 0, stream>>>(A, lda, W, ldw, bias, res, op, ldo, M, N, K);
    } else {
      if (epi == EPI_RESADD)
        gemm_bf16_kernel<false, EPI_RESADD><<<g, blk, 0, stream>>>(A, lda, W, ldw, bias, res, op, ldo, M, N, K);
      else
        gemm_bf16_kernel<false, EPI_NONE><<<g, blk, 0, stream>>>(A, lda, W, ldw, bias, res, op, ldo, M, N, K);
    }
  };

  // --- QKV projections (A = n1 inside comb, lda = 2D) ---
  gemm(comb, 2 * D, wq_b, D, bq, nullptr, qb, D, BS, D, D, EPI_NONE, true);
  gemm(comb, 2 * D, wk_b, D, bk, nullptr, kb, D, BS, D, D, EPI_NONE, true);
  gemm(comb, 2 * D, wv_b, D, bv, nullptr, vb, D, BS, D, D, EPI_NONE, true);

  // --- V transpose for PV matmul ---
  transpose_v_kernel<<<(BS * D) / 256, 256, 0, stream>>>(vb, vtb);

  // --- flash attention; writes ctx into comb[:, D:2D] ---
  {
    dim3 g(B * H, S / 128), blk(256);
    attn_kernel<<<g, blk, 0, stream>>>(qb, kb, vtb, comb);
  }

  // --- router query = [n1|ctx] @ Ws^T + bs ---
  gemm(comb, 2 * D, ws_b, 2 * D, bsb, nullptr, qryb, D, BS, D, 2 * D, EPI_NONE, true);

  // --- neuron scores = query @ emb^T ---
  gemm(qryb, D, emb_b, D, nullptr, nullptr, scf, NN, BS, NN, D, EPI_NONE, false);

  // --- top-k + softmax + gather ---
  router_kernel<<<BS, 256, 0, stream>>>(scf, emb, infob);

  // --- x_aug = n2 + info @ Wnp^T + bnp  (bf16 out) ---
  gemm(infob, D, wnp_b, D, bnp, n2f, xaugb, D, BS, D, D, EPI_RESADD, true);

  // --- h = gelu(x_aug @ Wup^T + bup)  (bf16 out) ---
  gemm(xaugb, D, wup_b, D, bup, nullptr, hb, DFF, BS, DFF, D, EPI_GELU, true);

  // --- out = x + h @ Wdown^T + bdown  (f32 out) ---
  gemm(hb, DFF, wdn_b, DFF, bdn, x, out, D, BS, D, DFF, EPI_RESADD, false);

  (void)in_sizes; (void)n_in; (void)out_size; (void)ws_size;
}
